// HybridPhysicsGNN_55241869361627
// MI455X (gfx1250) — matrix operand order repaired and checked by reading, not compiled, not run on hardware
//
#include <hip/hip_runtime.h>
#include <math.h>

// ---------------------------------------------------------------------------
// HybridPhysicsGNN forward for MI455X (gfx1250).
// Dense GEMMs -> v_wmma_f32_16x16x32_f16; weights pre-packed per launch into
// WMMA B-fragment order (f16) so the inner loop is pure b128 loads + wmma.
// One wave computes a 16x64 strip (A converted once, 4 WMMAs per K-step).
// Sparse edge/segment ops -> wave32 coalesced gather + f32 atomics.
// ---------------------------------------------------------------------------

typedef __attribute__((ext_vector_type(16))) _Float16 v16h;
typedef __attribute__((ext_vector_type(2)))  _Float16 v2h;
typedef __attribute__((ext_vector_type(8)))  float    v8f;

#define DD   64      // hidden dim
#define HH   4       // heads
#define HD   256     // H*D
#define BB   64      // graphs
#define LL   4       // layers

// ---- helpers --------------------------------------------------------------

__device__ __forceinline__ v2h pk2h(float lo, float hi) {
#if __has_builtin(__builtin_amdgcn_cvt_pkrtz)
  return __builtin_bit_cast(v2h, __builtin_amdgcn_cvt_pkrtz(lo, hi));
#else
  v2h r; r.x = (_Float16)lo; r.y = (_Float16)hi; return r;
#endif
}
__device__ __forceinline__ int f2ord(float f) {              // order-preserving
  int i = __float_as_int(f);
  return (i >= 0) ? i : (i ^ 0x7fffffff);
}
__device__ __forceinline__ float ord2f(int m) {
  return __int_as_float((m >= 0) ? m : (m ^ 0x7fffffff));
}
__device__ __forceinline__ float gelu_f(float x) {           // jax tanh approx
  float x3 = x * x * x;
  return 0.5f * x * (1.0f + tanhf(0.7978845608028654f * (x + 0.044715f * x3)));
}
#define NEG_INF_ORD 0x807FFFFFu  // f2ord(-inf) bit pattern

union HFrag { v16h v; v2h p[8]; };

// ---- weight repack: fp32 [K,M] -> f16 WMMA B-fragment order ----------------
// Wp[((k0/32)*(M/16) + tile)*512 + lane*16 + i], where for lane=(sub,l16):
//   element i maps to K = k0 + sub*16 + i, col = tile*16 + l16.
__global__ __launch_bounds__(256)
void repack_w_kernel(const float* __restrict__ W, _Float16* __restrict__ Wp,
                     int K, int M) {
  int idx = blockIdx.x * blockDim.x + threadIdx.x;   // total = K*M
  if (idx >= K * M) return;
  int i    = idx & 15;
  int lane = (idx >> 4) & 31;
  int frag = idx >> 9;                               // kb*(M/16) + tile
  int mt   = M >> 4;
  int kb   = frag / mt;
  int t    = frag - kb * mt;
  int k    = kb * 32 + (lane >> 4) * 16 + i;
  int col  = t * 16 + (lane & 15);
  Wp[idx] = (_Float16)W[(size_t)k * M + col];
}

// ---- WMMA GEMM: C = act(A[nxK] @ W[KxM] + bias) (+ residual) --------------
// One wave -> 16 rows x 64 cols (4 column tiles). 4 waves/block -> 64 rows.
// K % 32 == 0, M % 64 == 0. Wp is the fragment-packed f16 weight.
// Tail rows: A loads are clamped to the last valid row (harmless duplicate
// compute); stores are guarded, so any n_rows is correct with no inner-loop
// branches.

__global__ __launch_bounds__(128)
void gemm_f16_wmma_kernel(const float* __restrict__ A,
                          const _Float16* __restrict__ Wp,
                          const float* __restrict__ bias,
                          const float* __restrict__ residual,
                          float* __restrict__ C,
                          int n_rows, int K, int M, int act)
{
  const int wave = threadIdx.x >> 5;
  const int lane = threadIdx.x & 31;
  const int sub  = lane >> 4;          // half-wave (0/1)
  const int l16  = lane & 15;

  const int row_tiles = (n_rows + 15) >> 4;
  const int tile_m = blockIdx.x * 4 + wave;
  if (tile_m >= row_tiles) return;
  const int tb0  = blockIdx.y * 4;               // first of 4 column tiles
  const int colb = tb0 * 16;
  const int mt   = M >> 4;

  const int row_a = min(tile_m * 16 + l16, n_rows - 1);   // clamped A row
  const float* __restrict__ Arow = A + (size_t)row_a * (size_t)K;

  v8f acc[4] = {{}, {}, {}, {}};
  for (int k0 = 0; k0 < K; k0 += 32) {
    // A fragment (16x32 f16): lane(l16,sub) holds K = sub*16+[0..15]
    const float4* A4a = (const float4*)(Arow + k0 + sub * 16);
    const float4* A4b = (const float4*)(Arow + k0 + sub * 16 + 8);
    float4 a0 = A4a[0], a1 = A4a[1], a2 = A4b[0], a3 = A4b[1];
    HFrag af;
    af.p[0] = pk2h(a0.x, a0.y); af.p[1] = pk2h(a0.z, a0.w);
    af.p[2] = pk2h(a1.x, a1.y); af.p[3] = pk2h(a1.z, a1.w);
    af.p[4] = pk2h(a2.x, a2.y); af.p[5] = pk2h(a2.z, a2.w);
    af.p[6] = pk2h(a3.x, a3.y); af.p[7] = pk2h(a3.z, a3.w);

    // B fragments: contiguous 32B per lane from the packed weight.
    const _Float16* __restrict__ fb =
        Wp + (((size_t)(k0 >> 5) * mt + tb0) * 32 + lane) * 16;
    #pragma unroll
    for (int t = 0; t < 4; ++t) {
      v16h bf = *(const v16h*)(fb + (size_t)t * 512);
      acc[t] = __builtin_amdgcn_wmma_f32_16x16x32_f16(
          false, af.v, false, bf, (short)0, acc[t], false, false);
    }
  }

  #pragma unroll
  for (int t = 0; t < 4; ++t) {
    const int col = colb + t * 16 + l16;
    const float bv = bias ? bias[col] : 0.0f;
    #pragma unroll
    for (int r = 0; r < 8; ++r) {               // C element r -> row sub*8+r
      int row = tile_m * 16 + sub * 8 + r;
      if (row < n_rows) {
        float v = acc[t][r] + bv;
        if (act == 1)      v = (v >= 0.f) ? v : 0.2f * v;   // leaky
        else if (act == 2) v = gelu_f(v);                   // gelu
        size_t idx = (size_t)row * M + col;
        if (residual) v += residual[idx];
        C[idx] = v;
      }
    }
  }
}

// ---- small kernels --------------------------------------------------------

__global__ __launch_bounds__(256)
void fill_u32_kernel(unsigned int* p, unsigned int v, long n) {
  long i = (long)blockIdx.x * blockDim.x + threadIdx.x;
  if (i < n) p[i] = v;
}

__global__ __launch_bounds__(256)
void encoder_kernel(const float* __restrict__ x, const float* __restrict__ W,
                    const float* __restrict__ b, float* __restrict__ h, int n) {
  int i = blockIdx.x * blockDim.x + threadIdx.x;
  if (i >= n * DD) return;
  int node = i >> 6, d = i & 63;
  const float* xn = x + (size_t)node * 7;
  float acc = b[d];
  #pragma unroll
  for (int k = 0; k < 7; ++k) acc += xn[k] * W[k * DD + d];
  h[i] = acc;
}

// Pass A: per-edge attention logits + segment max (one wave per edge).
__global__ __launch_bounds__(256)
void edge_score_kernel(const int* __restrict__ ei, const float* __restrict__ xl,
                       const float* __restrict__ xr, const float* __restrict__ att,
                       float* __restrict__ ea, int* __restrict__ mord,
                       int Etot, int Ereal) {
  int edge = blockIdx.x * 8 + (threadIdx.x >> 5);
  if (edge >= Etot) return;
  int lane = threadIdx.x & 31;
  int s, d;
  if (edge < Ereal) { s = ei[edge]; d = ei[Ereal + edge]; }
  else              { s = edge - Ereal; d = s; }
  const float* xls = xl + (size_t)s * HD;
  const float* xrd = xr + (size_t)d * HD;
  float acc[HH] = {0.f, 0.f, 0.f, 0.f};
  #pragma unroll
  for (int j = 0; j < 8; ++j) {
    int idx = j * 32 + lane;                 // head = idx/64 = j>>1
    float v = xls[idx] + xrd[idx];
    v = (v >= 0.f) ? v : 0.2f * v;
    acc[j >> 1] += v * att[idx];
  }
  #pragma unroll
  for (int hh = 0; hh < HH; ++hh) {
    #pragma unroll
    for (int off = 16; off; off >>= 1) acc[hh] += __shfl_xor(acc[hh], off, 32);
  }
  if (lane == 0) {
    #pragma unroll
    for (int hh = 0; hh < HH; ++hh) {
      ea[(size_t)edge * HH + hh] = acc[hh];
      atomicMax(&mord[d * HH + hh], f2ord(acc[hh]));
    }
  }
}

// Pass B: a = exp(e - max), accumulate denominators.
__global__ __launch_bounds__(256)
void edge_softmax_kernel(const int* __restrict__ ei, const int* __restrict__ mord,
                         float* __restrict__ ea, float* __restrict__ denom,
                         int Etot, int Ereal) {
  int gid = blockIdx.x * blockDim.x + threadIdx.x;
  if (gid >= Etot * HH) return;
  int edge = gid >> 2, hh = gid & 3;
  int d = (edge < Ereal) ? ei[Ereal + edge] : (edge - Ereal);
  float a = __expf(ea[gid] - ord2f(mord[d * HH + hh]));
  ea[gid] = a;
  atomicAdd(&denom[d * HH + hh], a);
}

// Pass C: scatter normalized a * xl[src] into destination accumulator.
__global__ __launch_bounds__(256)
void edge_scatter_kernel(const int* __restrict__ ei, const float* __restrict__ xl,
                         const float* __restrict__ ea, const float* __restrict__ denom,
                         float* __restrict__ outacc, int Etot, int Ereal) {
  int edge = blockIdx.x * 8 + (threadIdx.x >> 5);
  if (edge >= Etot) return;
  int lane = threadIdx.x & 31;
  int s, d;
  if (edge < Ereal) { s = ei[edge]; d = ei[Ereal + edge]; }
  else              { s = edge - Ereal; d = s; }
  float an[HH];
  #pragma unroll
  for (int hh = 0; hh < HH; ++hh)
    an[hh] = ea[(size_t)edge * HH + hh] / (denom[(size_t)d * HH + hh] + 1e-16f);
  const float* xls = xl + (size_t)s * HD;
  float* od = outacc + (size_t)d * HD;
  #pragma unroll
  for (int j = 0; j < 8; ++j) {
    int idx = j * 32 + lane;
    atomicAdd(&od[idx], an[j >> 1] * xls[idx]);
  }
}

// head-mean + bias + residual + LayerNorm (one wave per node).
__global__ __launch_bounds__(256)
void gat_finalize_kernel(const float* __restrict__ h, const float* __restrict__ outacc,
                         const float* __restrict__ gb, const float* __restrict__ lg,
                         const float* __restrict__ lb, float* __restrict__ h2, int n) {
  int node = blockIdx.x * 8 + (threadIdx.x >> 5);
  if (node >= n) return;
  int lane = threadIdx.x & 31;
  const float* oa = outacc + (size_t)node * HD;
  const float* hn = h + (size_t)node * DD;
  float v[2];
  #pragma unroll
  for (int j = 0; j < 2; ++j) {
    int k = lane + j * 32;
    float s = 0.25f * (oa[k] + oa[64 + k] + oa[128 + k] + oa[192 + k]) + gb[k];
    v[j] = hn[k] + s;
  }
  float sum = v[0] + v[1];
  #pragma unroll
  for (int off = 16; off; off >>= 1) sum += __shfl_xor(sum, off, 32);
  float mu = sum * (1.0f / 64.0f);
  float var = (v[0] - mu) * (v[0] - mu) + (v[1] - mu) * (v[1] - mu);
  #pragma unroll
  for (int off = 16; off; off >>= 1) var += __shfl_xor(var, off, 32);
  float rs = rsqrtf(var * (1.0f / 64.0f) + 1e-5f);
  #pragma unroll
  for (int j = 0; j < 2; ++j) {
    int k = lane + j * 32;
    h2[(size_t)node * DD + k] = (v[j] - mu) * rs * lg[k] + lb[k];
  }
}

__global__ __launch_bounds__(256)
void count_kernel(const int* __restrict__ batch, int* __restrict__ cnt, int n) {
  int i = blockIdx.x * blockDim.x + threadIdx.x;
  if (i < n) atomicAdd(&cnt[batch[i]], 1);
}

__global__ __launch_bounds__(256)
void pool_scatter_kernel(const float* __restrict__ lat, const int* __restrict__ batch,
                         float* __restrict__ gsum, int* __restrict__ gmaxo, int n) {
  int i = blockIdx.x * blockDim.x + threadIdx.x;
  if (i >= n * DD) return;
  int node = i >> 6, k = i & 63;
  int b = batch[node];
  float v = lat[i];
  atomicAdd(&gsum[b * DD + k], v);
  atomicMax(&gmaxo[b * DD + k], f2ord(v));
}

__global__ __launch_bounds__(256)
void glbuild_kernel(const float* __restrict__ gsum, const int* __restrict__ gmaxo,
                    const int* __restrict__ cnt, float* __restrict__ gl) {
  int i = blockIdx.x * blockDim.x + threadIdx.x;
  if (i >= BB * DD) return;
  int b = i >> 6, k = i & 63;
  float c = (float)max(cnt[b], 1);
  gl[b * (2 * DD) + k]      = gsum[i] / c;
  gl[b * (2 * DD) + DD + k] = ord2f(gmaxo[i]);
}

__global__ __launch_bounds__(256)
void ctx_add_kernel(const float* __restrict__ ctx, const int* __restrict__ batch,
                    float* __restrict__ h, int n) {
  int i = blockIdx.x * blockDim.x + threadIdx.x;
  if (i >= n * DD) return;
  int node = i >> 6, k = i & 63;
  h[i] += ctx[batch[node] * DD + k];
}

__global__ __launch_bounds__(256)
void decoder2_kernel(const float* __restrict__ t, const float* __restrict__ W2,
                     const float* __restrict__ b2, float* __restrict__ out, int n) {
  int node = blockIdx.x * blockDim.x + threadIdx.x;
  if (node >= n) return;
  const float* tn = t + (size_t)node * DD;
  float a0 = b2[0], a1 = b2[1];
  #pragma unroll
  for (int k = 0; k < DD; ++k) {
    float tv = tn[k];
    a0 += tv * W2[k * 2];
    a1 += tv * W2[k * 2 + 1];
  }
  out[node * 2]     = a0;
  out[node * 2 + 1] = a1;
}

// ---- host-side orchestration ---------------------------------------------

static inline void fill_u32(void* p, unsigned int v, long cnt, hipStream_t s) {
  fill_u32_kernel<<<(unsigned)((cnt + 255) / 256), 256, 0, s>>>((unsigned int*)p, v, cnt);
}
static inline void repack_w(const float* W, _Float16* Wp, int K, int M,
                            hipStream_t s) {
  long total = (long)K * M;
  repack_w_kernel<<<(unsigned)((total + 255) / 256), 256, 0, s>>>(W, Wp, K, M);
}
static inline void launch_gemm(const float* A, const _Float16* Wp, const float* bias,
                               const float* res, float* C, int n_rows, int K, int M,
                               int act, hipStream_t s) {
  int rt = (n_rows + 15) / 16;
  dim3 grid((rt + 3) / 4, M / 64);
  gemm_f16_wmma_kernel<<<grid, 128, 0, s>>>(A, Wp, bias, res, C, n_rows, K, M, act);
}

extern "C" void kernel_launch(void* const* d_in, const int* in_sizes, int n_in,
                              void* d_out, int out_size, void* d_ws, size_t ws_size,
                              hipStream_t stream) {
  const float* x       = (const float*)d_in[0];
  const int*   ei      = (const int*)d_in[1];
  const int*   batch   = (const int*)d_in[2];
  const float* enc_W   = (const float*)d_in[3];
  const float* enc_b   = (const float*)d_in[4];
  const float* gat_Wl  = (const float*)d_in[5];
  const float* gat_Wr  = (const float*)d_in[6];
  const float* gat_att = (const float*)d_in[7];
  const float* gat_b   = (const float*)d_in[8];
  const float* ln_g    = (const float*)d_in[9];
  const float* ln_b    = (const float*)d_in[10];
  const float* ffn_W1  = (const float*)d_in[11];
  const float* ffn_b1  = (const float*)d_in[12];
  const float* ffn_W2  = (const float*)d_in[13];
  const float* ffn_b2  = (const float*)d_in[14];
  const float* n2g_W1  = (const float*)d_in[15];
  const float* n2g_b1  = (const float*)d_in[16];
  const float* n2g_W2  = (const float*)d_in[17];
  const float* n2g_b2  = (const float*)d_in[18];
  const float* g2n_W1  = (const float*)d_in[19];
  const float* g2n_b1  = (const float*)d_in[20];
  const float* g2n_W2  = (const float*)d_in[21];
  const float* g2n_b2  = (const float*)d_in[22];
  const float* dec_W1  = (const float*)d_in[23];
  const float* dec_b1  = (const float*)d_in[24];
  const float* dec_W2  = (const float*)d_in[25];
  const float* dec_b2  = (const float*)d_in[26];

  const int n  = in_sizes[2];        // N nodes
  const int E  = in_sizes[1] / 2;    // edges
  const int Et = E + n;              // + self loops

  // Workspace carve (256B aligned slabs).
  size_t off = 0;
  char* base = (char*)d_ws;
  auto carve = [&](size_t bytes) -> char* {
    char* p = base + off;
    off = (off + bytes + 255) & ~(size_t)255;
    return p;
  };
  float* h      = (float*)carve((size_t)n * DD * 4);
  float* h2     = (float*)carve((size_t)n * DD * 4);
  float* xl     = (float*)carve((size_t)n * HD * 4);
  float* xr     = (float*)carve((size_t)n * HD * 4);
  float* ea     = (float*)carve((size_t)Et * HH * 4);
  int*   mord   = (int*)  carve((size_t)n * HH * 4);
  float* denom  = (float*)carve((size_t)n * HH * 4);
  float* outacc = (float*)carve((size_t)n * HD * 4);
  float* t1     = (float*)carve((size_t)n * 2 * DD * 4);   // FFN hidden; reused as temp
  float* lat    = (float*)carve((size_t)n * DD * 4);
  float* gsum   = (float*)carve((size_t)BB * DD * 4);
  int*   gmaxo  = (int*)  carve((size_t)BB * DD * 4);
  int*   cnt    = (int*)  carve((size_t)BB * 4);
  float* gl     = (float*)carve((size_t)BB * 2 * DD * 4);
  float* ctxt   = (float*)carve((size_t)BB * DD * 4);
  float* ctx    = (float*)carve((size_t)BB * DD * 4);
  // Fragment-packed f16 weights (one slab per weight family, per-layer slices).
  _Float16* p_gatWl = (_Float16*)carve((size_t)LL * DD * HD * 2);
  _Float16* p_gatWr = (_Float16*)carve((size_t)LL * DD * HD * 2);
  _Float16* p_ffnW1 = (_Float16*)carve((size_t)LL * DD * 2 * DD * 2);
  _Float16* p_ffnW2 = (_Float16*)carve((size_t)LL * 2 * DD * DD * 2);
  _Float16* p_n2gW1 = (_Float16*)carve((size_t)LL * DD * DD * 2);
  _Float16* p_n2gW2 = (_Float16*)carve((size_t)LL * DD * DD * 2);
  _Float16* p_g2nW1 = (_Float16*)carve((size_t)LL * 2 * DD * DD * 2);
  _Float16* p_g2nW2 = (_Float16*)carve((size_t)LL * DD * DD * 2);
  _Float16* p_decW1 = (_Float16*)carve((size_t)DD * DD * 2);
  (void)ws_size; (void)n_in; (void)out_size;

  // ---- one-time per launch: pack all GEMM weights into fragment order ----
  for (int l = 0; l < LL; ++l) {
    repack_w(gat_Wl + (size_t)l * DD * HD,        p_gatWl + (size_t)l * DD * HD,        DD,     HD, stream);
    repack_w(gat_Wr + (size_t)l * DD * HD,        p_gatWr + (size_t)l * DD * HD,        DD,     HD, stream);
    repack_w(ffn_W1 + (size_t)l * DD * 2 * DD,    p_ffnW1 + (size_t)l * DD * 2 * DD,    DD, 2 * DD, stream);
    repack_w(ffn_W2 + (size_t)l * 2 * DD * DD,    p_ffnW2 + (size_t)l * 2 * DD * DD,    2 * DD, DD, stream);
    repack_w(n2g_W1 + (size_t)l * DD * DD,        p_n2gW1 + (size_t)l * DD * DD,        DD,     DD, stream);
    repack_w(n2g_W2 + (size_t)l * DD * DD,        p_n2gW2 + (size_t)l * DD * DD,        DD,     DD, stream);
    repack_w(g2n_W1 + (size_t)l * 2 * DD * DD,    p_g2nW1 + (size_t)l * 2 * DD * DD,    2 * DD, DD, stream);
    repack_w(g2n_W2 + (size_t)l * DD * DD,        p_g2nW2 + (size_t)l * DD * DD,        DD,     DD, stream);
  }
  repack_w(dec_W1, p_decW1, DD, DD, stream);

  // Graph node counts (batch assignment is a fixed input).
  fill_u32(cnt, 0u, BB, stream);
  count_kernel<<<(n + 255) / 256, 256, 0, stream>>>(batch, cnt, n);

  // Encoder: h = x @ enc_W + enc_b   (K=7, scalar path)
  encoder_kernel<<<((long)n * DD + 255) / 256, 256, 0, stream>>>(x, enc_W, enc_b, h, n);

  for (int l = 0; l < LL; ++l) {
    // ---- GATv2 transforms (WMMA) ----
    launch_gemm(h, p_gatWl + (size_t)l * DD * HD, nullptr, nullptr, xl, n, DD, HD, 0, stream);
    launch_gemm(h, p_gatWr + (size_t)l * DD * HD, nullptr, nullptr, xr, n, DD, HD, 0, stream);

    // ---- segment softmax over destinations ----
    fill_u32(mord, NEG_INF_ORD, (long)n * HH, stream);
    fill_u32(denom, 0u, (long)n * HH, stream);
    fill_u32(outacc, 0u, (long)n * HD, stream);
    edge_score_kernel<<<(Et + 7) / 8, 256, 0, stream>>>(
        ei, xl, xr, gat_att + (size_t)l * HD, ea, mord, Et, E);
    edge_softmax_kernel<<<((long)Et * HH + 255) / 256, 256, 0, stream>>>(
        ei, mord, ea, denom, Et, E);
    edge_scatter_kernel<<<(Et + 7) / 8, 256, 0, stream>>>(
        ei, xl, ea, denom, outacc, Et, E);

    // ---- head mean + residual + LayerNorm -> h2 ----
    gat_finalize_kernel<<<(n + 7) / 8, 256, 0, stream>>>(
        h, outacc, gat_b + (size_t)l * DD, ln_g + (size_t)l * DD,
        ln_b + (size_t)l * DD, h2, n);

    // ---- FFN: h = h2 + leaky(h2@W1+b1)@W2+b2 ----
    launch_gemm(h2, p_ffnW1 + (size_t)l * DD * 2 * DD, ffn_b1 + (size_t)l * (2 * DD),
                nullptr, t1, n, DD, 2 * DD, 1, stream);
    launch_gemm(t1, p_ffnW2 + (size_t)l * 2 * DD * DD, ffn_b2 + (size_t)l * DD,
                h2, h, n, 2 * DD, DD, 0, stream);

    // ---- GlobalContextLayer ----
    launch_gemm(h, p_n2gW1 + (size_t)l * DD * DD, n2g_b1 + (size_t)l * DD,
                nullptr, t1, n, DD, DD, 2, stream);
    launch_gemm(t1, p_n2gW2 + (size_t)l * DD * DD, n2g_b2 + (size_t)l * DD,
                nullptr, lat, n, DD, DD, 0, stream);
    fill_u32(gsum, 0u, BB * DD, stream);
    fill_u32(gmaxo, NEG_INF_ORD, BB * DD, stream);
    pool_scatter_kernel<<<((long)n * DD + 255) / 256, 256, 0, stream>>>(
        lat, batch, gsum, gmaxo, n);
    glbuild_kernel<<<(BB * DD + 255) / 256, 256, 0, stream>>>(gsum, gmaxo, cnt, gl);
    launch_gemm(gl, p_g2nW1 + (size_t)l * 2 * DD * DD, g2n_b1 + (size_t)l * DD,
                nullptr, ctxt, BB, 2 * DD, DD, 2, stream);
    launch_gemm(ctxt, p_g2nW2 + (size_t)l * DD * DD, g2n_b2 + (size_t)l * DD,
                nullptr, ctx, BB, DD, DD, 0, stream);
    ctx_add_kernel<<<((long)n * DD + 255) / 256, 256, 0, stream>>>(ctx, batch, h, n);
  }

  // ---- decoder: out = gelu(h@dec_W1+b1)@dec_W2+b2  (second GEMM M=2 scalar) ----
  launch_gemm(h, p_decW1, dec_b1, nullptr, t1, n, DD, DD, 2, stream);
  decoder2_kernel<<<(n + 255) / 256, 256, 0, stream>>>(t1, dec_W2, dec_b2,
                                                       (float*)d_out, n);
}
